// TransformerBlock_34746285425444
// MI455X (gfx1250) — matrix operand order repaired and checked by reading, compiled, not run on hardware
//
#include <hip/hip_runtime.h>
#include <hip/hip_bf16.h>
#include <math.h>

#define D_MODEL   1024
#define N_HEADS   16
#define HEAD_DIM  64
#define D_FF      4096
#define BATCH     2
#define SEQ       2048
#define MROWS     (BATCH*SEQ)
#define EPS       1e-5f
#define LN10K     9.210340371976184f   /* ln(10000) */

typedef __attribute__((ext_vector_type(16))) __bf16 bf16x16;
typedef __attribute__((ext_vector_type(8)))  __bf16 bf16x8;
typedef __attribute__((ext_vector_type(8)))  float  f32x8;

static __device__ __forceinline__ __bf16 f2bf(float f) {
  unsigned int u = __builtin_bit_cast(unsigned int, f);
  u += 0x7fffu + ((u >> 16) & 1u);            // round-to-nearest-even
  unsigned short h = (unsigned short)(u >> 16);
  return __builtin_bit_cast(__bf16, h);
}

// ---------------- WMMA fragment loaders (wave32, ISA 7.12.2 layouts) -------
// A (16x32 bf16): lane: row m = l&15, hi = l>>4; element e -> K = koff + (e<8?0:16) + hi*8 + (e&7)
static __device__ __forceinline__ bf16x16 load_a(const __bf16* row, int koff, int hi) {
  bf16x8 lo  = *(const bf16x8*)(row + koff + hi*8);
  bf16x8 hi8 = *(const bf16x8*)(row + koff + 16 + hi*8);
  bf16x16 r;
#pragma unroll
  for (int i = 0; i < 8; ++i) { r[i] = lo[i]; r[8+i] = hi8[i]; }
  return r;
}
// B (32x16 bf16): lane: col n = l&15, hi = l>>4; element e -> K = koff + (e<8?0:16) + hi*8 + (e&7)
static __device__ __forceinline__ bf16x16 load_b(const __bf16* col, int koff, int hi) {
  const __bf16* p = col + koff + hi*8;
  bf16x8 lo  = *(const bf16x8*)(p);
  bf16x8 hi8 = *(const bf16x8*)(p + 16);
  bf16x16 r;
#pragma unroll
  for (int i = 0; i < 8; ++i) { r[i] = lo[i]; r[8+i] = hi8[i]; }
  return r;
}
static __device__ __forceinline__ f32x8 wmma_bf16(bf16x16 a, bf16x16 b, f32x8 c) {
  return __builtin_amdgcn_wmma_f32_16x16x32_bf16(false, a, false, b, (short)0, c, false, false);
}

// ---------------- fp32 -> bf16 convert (weights) ---------------------------
__global__ void f2bf_k(const float* __restrict__ in, __bf16* __restrict__ out, int n) {
  for (int i = blockIdx.x * blockDim.x + threadIdx.x; i < n; i += gridDim.x * blockDim.x)
    out[i] = f2bf(in[i]);
}

// ---------------- RMSNorm: f32 in, bf16 out --------------------------------
__global__ void rmsnorm_k(const float* __restrict__ x, const float* __restrict__ w,
                          __bf16* __restrict__ out) {
  const int row = blockIdx.x;
  const float* xr = x + (size_t)row * D_MODEL;
  __shared__ float red[256];
  float ss = 0.f;
  for (int i = threadIdx.x; i < D_MODEL; i += 256) { float v = xr[i]; ss += v * v; }
  red[threadIdx.x] = ss; __syncthreads();
  for (int st = 128; st > 0; st >>= 1) {
    if (threadIdx.x < st) red[threadIdx.x] += red[threadIdx.x + st];
    __syncthreads();
  }
  const float rinv = rsqrtf(red[0] / (float)D_MODEL + EPS);
  __bf16* orow = out + (size_t)row * D_MODEL;
  for (int i = threadIdx.x; i < D_MODEL; i += 256) orow[i] = f2bf(xr[i] * rinv * w[i]);
}

// ---------------- GEMM: C[M,N] = A[M,K] * B[N,K]^T -------------------------
// 32x32 macro-tile per wave (2x2 of 16x16), register double-buffered K loop.
enum { EPI_ROPE = 0, EPI_VT = 1, EPI_RESID = 2, EPI_ADD = 3 };

template <int EPI>
__global__ void gemm_wmma_k(const __bf16* __restrict__ A, const __bf16* __restrict__ B,
                            int M, int N, int K,
                            float* __restrict__ outF, const float* __restrict__ resid,
                            __bf16* __restrict__ outB, const int* __restrict__ tp) {
  const int lane = threadIdx.x & 31, wave = threadIdx.x >> 5;
  const int m = lane & 15, hi = lane >> 4;
  const int tilesN = N >> 5;
  const int t = blockIdx.x * (blockDim.x >> 5) + wave;
  if (t >= (M >> 5) * tilesN) return;
  const int tm = t / tilesN, tn = t % tilesN;
  const __bf16* arow0 = A + (size_t)(tm * 32 + m) * K;
  const __bf16* arow1 = arow0 + (size_t)16 * K;
  const __bf16* brow0 = B + (size_t)(tn * 32 + m) * K;
  const __bf16* brow1 = brow0 + (size_t)16 * K;

  f32x8 c00 = {}, c01 = {}, c10 = {}, c11 = {};
  bf16x16 a0 = load_a(arow0, 0, hi), a1 = load_a(arow1, 0, hi);
  bf16x16 b0 = load_b(brow0, 0, hi), b1 = load_b(brow1, 0, hi);
  const int iters = K >> 5;
  for (int i = 0; i < iters - 1; ++i) {
    const int kn = (i + 1) << 5;
    if (kn + 128 < K) {
      __builtin_prefetch(arow0 + kn + 128, 0, 3);
      __builtin_prefetch(brow0 + kn + 128, 0, 3);
    }
    // issue next K-step's loads before consuming current fragments
    const bf16x16 a0n = load_a(arow0, kn, hi), a1n = load_a(arow1, kn, hi);
    const bf16x16 b0n = load_b(brow0, kn, hi), b1n = load_b(brow1, kn, hi);
    c00 = wmma_bf16(a0, b0, c00);
    c01 = wmma_bf16(a0, b1, c01);
    c10 = wmma_bf16(a1, b0, c10);
    c11 = wmma_bf16(a1, b1, c11);
    a0 = a0n; a1 = a1n; b0 = b0n; b1 = b1n;
  }
  c00 = wmma_bf16(a0, b0, c00);
  c01 = wmma_bf16(a0, b1, c01);
  c10 = wmma_bf16(a1, b0, c10);
  c11 = wmma_bf16(a1, b1, c11);

  auto epi_tile = [&](const f32x8& c, int rowb, int colb) {
    const int col = colb + m;
#pragma unroll
    for (int v = 0; v < 8; ++v) {
      const int row = rowb + v + 8 * hi;
      float acc = c[v];
      if constexpr (EPI == EPI_RESID) {
        outF[(size_t)row * N + col] = resid[(size_t)row * N + col] + acc;
      } else if constexpr (EPI == EPI_ADD) {
        outF[(size_t)row * N + col] += acc;
      } else if constexpr (EPI == EPI_VT) {
        const int b = row / SEQ, s = row - b * SEQ;
        const int h = col >> 6, hd = col & 63;
        outB[(((size_t)(b * N_HEADS + h)) * HEAD_DIM + hd) * SEQ + s] = f2bf(acc);
      } else {  // EPI_ROPE: pair partner (hd^1) lives in lane^1, same VGPR
        const float other = __shfl_xor(acc, 1);
        const int b = row / SEQ, s = row - b * SEQ;
        const int h = col >> 6, hd = col & 63;
        const float pos = (float)tp[s];
        const float th = pos * __expf(-((float)(hd & ~1)) * (LN10K / (float)HEAD_DIM));
        float sn, cs; __sincosf(th, &sn, &cs);
        const float res = ((hd & 1) == 0) ? (acc * cs - other * sn) : (other * sn + acc * cs);
        outB[(((size_t)(b * N_HEADS + h)) * SEQ + s) * HEAD_DIM + hd] = f2bf(res);
      }
    }
  };
  epi_tile(c00, tm * 32,      tn * 32);
  epi_tile(c01, tm * 32,      tn * 32 + 16);
  epi_tile(c10, tm * 32 + 16, tn * 32);
  epi_tile(c11, tm * 32 + 16, tn * 32 + 16);
}

// ---------------- Dual GEMM + SwiGLU: act = silu(A*W1^T) * (A*W3^T) --------
// 32 rows x 16 cols per wave; W1/W3 B-fragments shared across both row tiles.
__global__ void gemm_dual_silu_k(const __bf16* __restrict__ A,
                                 const __bf16* __restrict__ B1, const __bf16* __restrict__ B2,
                                 int M, int N, int K, __bf16* __restrict__ outB) {
  const int lane = threadIdx.x & 31, wave = threadIdx.x >> 5;
  const int m = lane & 15, hi = lane >> 4;
  const int tilesN = N >> 4;
  const int t = blockIdx.x * (blockDim.x >> 5) + wave;
  if (t >= (M >> 5) * tilesN) return;
  const int tm = t / tilesN, tn = t % tilesN;
  const __bf16* arow0 = A  + (size_t)(tm * 32 + m) * K;
  const __bf16* arow1 = arow0 + (size_t)16 * K;
  const __bf16* b1row = B1 + (size_t)(tn * 16 + m) * K;
  const __bf16* b2row = B2 + (size_t)(tn * 16 + m) * K;

  f32x8 cg0 = {}, cg1 = {}, cu0 = {}, cu1 = {};
  bf16x16 a0 = load_a(arow0, 0, hi), a1 = load_a(arow1, 0, hi);
  bf16x16 bg = load_b(b1row, 0, hi), bu = load_b(b2row, 0, hi);
  const int iters = K >> 5;
  for (int i = 0; i < iters - 1; ++i) {
    const int kn = (i + 1) << 5;
    if (kn + 128 < K) __builtin_prefetch(arow0 + kn + 128, 0, 3);
    const bf16x16 a0n = load_a(arow0, kn, hi), a1n = load_a(arow1, kn, hi);
    const bf16x16 bgn = load_b(b1row, kn, hi), bun = load_b(b2row, kn, hi);
    cg0 = wmma_bf16(a0, bg, cg0);
    cg1 = wmma_bf16(a1, bg, cg1);
    cu0 = wmma_bf16(a0, bu, cu0);
    cu1 = wmma_bf16(a1, bu, cu1);
    a0 = a0n; a1 = a1n; bg = bgn; bu = bun;
  }
  cg0 = wmma_bf16(a0, bg, cg0);
  cg1 = wmma_bf16(a1, bg, cg1);
  cu0 = wmma_bf16(a0, bu, cu0);
  cu1 = wmma_bf16(a1, bu, cu1);

  const int col = tn * 16 + m;
#pragma unroll
  for (int v = 0; v < 8; ++v) {
    const int r0 = tm * 32 + v + 8 * hi;
    const int r1 = r0 + 16;
    const float g0 = cg0[v], u0 = cu0[v];
    const float g1 = cg1[v], u1 = cu1[v];
    outB[(size_t)r0 * N + col] = f2bf(g0 * (1.f / (1.f + __expf(-g0))) * u0);
    outB[(size_t)r1 * N + col] = f2bf(g1 * (1.f / (1.f + __expf(-g1))) * u1);
  }
}

// ---------------- Flash attention (causal), wave32 WMMA --------------------
// Q,K: (b,h,s,64) bf16 (RoPE applied).  Vt: (b,h,64,s) bf16.  out: (b,s,1024) bf16.
__global__ void attn_k(const __bf16* __restrict__ Q, const __bf16* __restrict__ Km,
                       const __bf16* __restrict__ Vt, __bf16* __restrict__ out) {
  __shared__ __align__(16) __bf16 plds[8][16][32];
  const int lane = threadIdx.x & 31, w = threadIdx.x >> 5;
  const int m = lane & 15, hi = lane >> 4;
  const int bh = blockIdx.y, b = bh >> 4, h = bh & 15;
  const int q0 = blockIdx.x * 128 + w * 16;

  const __bf16* qrow = Q + ((size_t)bh * SEQ + q0 + m) * HEAD_DIM;
  const bf16x16 a0 = load_a(qrow, 0, hi);
  const bf16x16 a1 = load_a(qrow, 32, hi);

  f32x8 o0 = {}, o1 = {}, o2 = {}, o3 = {};
  float Mr[8], Lr[8];
#pragma unroll
  for (int v = 0; v < 8; ++v) { Mr[v] = -3.0e38f; Lr[v] = 0.f; }

  const int kend = q0 + 15;
  for (int kb = 0; kb <= kend; kb += 32) {
    const __bf16* k0row = Km + ((size_t)bh * SEQ + kb + m) * HEAD_DIM;
    const __bf16* k1row = Km + ((size_t)bh * SEQ + kb + 16 + m) * HEAD_DIM;
    f32x8 s0 = {}, s1 = {};
    s0 = wmma_bf16(a0, load_b(k0row, 0, hi), s0);
    s0 = wmma_bf16(a1, load_b(k0row, 32, hi), s0);
    s1 = wmma_bf16(a0, load_b(k1row, 0, hi), s1);
    s1 = wmma_bf16(a1, load_b(k1row, 32, hi), s1);

    float alpha[8];
#pragma unroll
    for (int v = 0; v < 8; ++v) {
      const int qi = q0 + v + 8 * hi;
      float x0 = s0[v] * 0.125f; if (kb + m > qi)      x0 = -1.0e30f;
      float x1 = s1[v] * 0.125f; if (kb + 16 + m > qi) x1 = -1.0e30f;
      // row reductions: a row lives across 16 lanes of a half-wave in one VGPR
      float mx = fmaxf(x0, x1);
      mx = fmaxf(mx, __shfl_xor(mx, 1)); mx = fmaxf(mx, __shfl_xor(mx, 2));
      mx = fmaxf(mx, __shfl_xor(mx, 4)); mx = fmaxf(mx, __shfl_xor(mx, 8));
      const float Mn = fmaxf(Mr[v], mx);
      const float al = __expf(Mr[v] - Mn);
      const float p0 = __expf(x0 - Mn), p1 = __expf(x1 - Mn);
      float rs = p0 + p1;
      rs += __shfl_xor(rs, 1); rs += __shfl_xor(rs, 2);
      rs += __shfl_xor(rs, 4); rs += __shfl_xor(rs, 8);
      Lr[v] = Lr[v] * al + rs; Mr[v] = Mn; alpha[v] = al;
      // stage P (C-layout -> row-major) for the A-fragment reload
      plds[w][v + 8 * hi][m]      = f2bf(p0);
      plds[w][v + 8 * hi][16 + m] = f2bf(p1);
    }
#pragma unroll
    for (int v = 0; v < 8; ++v) {
      o0[v] *= alpha[v]; o1[v] *= alpha[v]; o2[v] *= alpha[v]; o3[v] *= alpha[v];
    }
    const bf16x16 pa = load_a(&plds[w][m][0], 0, hi);  // intra-wave LDS: in-order, no barrier
    const size_t vbase = (size_t)bh * HEAD_DIM;
    o0 = wmma_bf16(pa, load_b(Vt + (vbase +  0 + m) * SEQ + kb, 0, hi), o0);
    o1 = wmma_bf16(pa, load_b(Vt + (vbase + 16 + m) * SEQ + kb, 0, hi), o1);
    o2 = wmma_bf16(pa, load_b(Vt + (vbase + 32 + m) * SEQ + kb, 0, hi), o2);
    o3 = wmma_bf16(pa, load_b(Vt + (vbase + 48 + m) * SEQ + kb, 0, hi), o3);
  }

#pragma unroll
  for (int v = 0; v < 8; ++v) {
    const float inv = 1.0f / Lr[v];
    const size_t rowoff = ((size_t)b * SEQ + q0 + v + 8 * hi) * D_MODEL + (size_t)h * HEAD_DIM;
    out[rowoff +  0 + m] = f2bf(o0[v] * inv);
    out[rowoff + 16 + m] = f2bf(o1[v] * inv);
    out[rowoff + 32 + m] = f2bf(o2[v] * inv);
    out[rowoff + 48 + m] = f2bf(o3[v] * inv);
  }
}

// ---------------------------------------------------------------------------
extern "C" void kernel_launch(void* const* d_in, const int* in_sizes, int n_in,
                              void* d_out, int out_size, void* d_ws, size_t ws_size,
                              hipStream_t stream) {
  (void)in_sizes; (void)n_in; (void)out_size; (void)ws_size;
  const float* x   = (const float*)d_in[0];
  const float* wq  = (const float*)d_in[1];
  const float* wk  = (const float*)d_in[2];
  const float* wv  = (const float*)d_in[3];
  const float* wo  = (const float*)d_in[4];
  const float* ln1 = (const float*)d_in[5];
  const float* ln2 = (const float*)d_in[6];
  const float* w1  = (const float*)d_in[7];
  const float* w2  = (const float*)d_in[8];
  const float* w3  = (const float*)d_in[9];
  const int*   tp  = (const int*)d_in[10];
  float* out = (float*)d_out;

  const size_t W_D2 = (size_t)D_MODEL * D_MODEL;  // 1M elems
  const size_t W_FF = (size_t)D_FF * D_MODEL;     // 4M elems
  char* ws = (char*)d_ws;
  size_t o = 0;
  auto alloc = [&](size_t bytes) { size_t r = o; o += (bytes + 255) & ~(size_t)255; return r; };
  __bf16* wqb   = (__bf16*)(ws + alloc(W_D2 * 2));
  __bf16* wkb   = (__bf16*)(ws + alloc(W_D2 * 2));
  __bf16* wvb   = (__bf16*)(ws + alloc(W_D2 * 2));
  __bf16* wob   = (__bf16*)(ws + alloc(W_D2 * 2));
  __bf16* w1b   = (__bf16*)(ws + alloc(W_FF * 2));
  __bf16* w3b   = (__bf16*)(ws + alloc(W_FF * 2));
  __bf16* w2b   = (__bf16*)(ws + alloc(W_FF * 2));
  __bf16* hb    = (__bf16*)(ws + alloc((size_t)MROWS * D_MODEL * 2));
  __bf16* qb    = (__bf16*)(ws + alloc((size_t)MROWS * D_MODEL * 2));
  __bf16* kbuf  = (__bf16*)(ws + alloc((size_t)MROWS * D_MODEL * 2));
  __bf16* vtb   = (__bf16*)(ws + alloc((size_t)MROWS * D_MODEL * 2));
  __bf16* attnb = (__bf16*)(ws + alloc((size_t)MROWS * D_MODEL * 2));
  __bf16* actb  = qb;  // 32MB FFN activation aliases dead q/k/vt/attn buffers

  const dim3 blk(256);
  // weights -> bf16 (once per launch; deterministic)
  f2bf_k<<<1024, blk, 0, stream>>>(wq, wqb, (int)W_D2);
  f2bf_k<<<1024, blk, 0, stream>>>(wk, wkb, (int)W_D2);
  f2bf_k<<<1024, blk, 0, stream>>>(wv, wvb, (int)W_D2);
  f2bf_k<<<1024, blk, 0, stream>>>(wo, wob, (int)W_D2);
  f2bf_k<<<2048, blk, 0, stream>>>(w1, w1b, (int)W_FF);
  f2bf_k<<<2048, blk, 0, stream>>>(w3, w3b, (int)W_FF);
  f2bf_k<<<2048, blk, 0, stream>>>(w2, w2b, (int)W_FF);

  rmsnorm_k<<<MROWS, blk, 0, stream>>>(x, ln1, hb);

  // 32x32 macro-tiles, 8 waves/block
  const int blocksD  = (MROWS / 32) * (D_MODEL / 32) / 8;  // 512
  const int blocksFF = (MROWS / 32) * (D_FF / 16) / 8;     // 4096
  gemm_wmma_k<EPI_ROPE><<<blocksD, blk, 0, stream>>>(hb, wqb, MROWS, D_MODEL, D_MODEL,
                                                     nullptr, nullptr, qb, tp);
  gemm_wmma_k<EPI_ROPE><<<blocksD, blk, 0, stream>>>(hb, wkb, MROWS, D_MODEL, D_MODEL,
                                                     nullptr, nullptr, kbuf, tp);
  gemm_wmma_k<EPI_VT><<<blocksD, blk, 0, stream>>>(hb, wvb, MROWS, D_MODEL, D_MODEL,
                                                   nullptr, nullptr, vtb, tp);

  attn_k<<<dim3(SEQ / 128, BATCH * N_HEADS), blk, 0, stream>>>(qb, kbuf, vtb, attnb);

  gemm_wmma_k<EPI_RESID><<<blocksD, blk, 0, stream>>>(attnb, wob, MROWS, D_MODEL, D_MODEL,
                                                      out, x, nullptr, nullptr);

  rmsnorm_k<<<MROWS, blk, 0, stream>>>(out, ln2, hb);

  gemm_dual_silu_k<<<blocksFF, blk, 0, stream>>>(hb, w1b, w3b, MROWS, D_FF, D_MODEL, actb);

  gemm_wmma_k<EPI_ADD><<<blocksD, blk, 0, stream>>>(actb, w2b, MROWS, D_MODEL, D_FF,
                                                    out, nullptr, nullptr, nullptr);
}
